// GINExpert_70531952935576
// MI455X (gfx1250) — matrix operand order repaired and checked by reading, not compiled, not run on hardware
//
#include <hip/hip_runtime.h>
#include <hip/hip_bf16.h>
#include <math.h>

// ---------------------------------------------------------------------------
// Types for CDNA5 WMMA (wave32): v_wmma_f32_16x16x32_bf16
// ---------------------------------------------------------------------------
typedef __bf16 bf16_t;
typedef __attribute__((ext_vector_type(8)))  __bf16 v8bf;
typedef __attribute__((ext_vector_type(16))) __bf16 v16bf;
typedef __attribute__((ext_vector_type(8)))  float  v8f;

__device__ __forceinline__ float elu1(float v) {
    return v > 0.0f ? v : (expf(v) - 1.0f);
}

// ---------------------------------------------------------------------------
// Weight transpose + convert: W (K x N, f32, row-major) -> Wt (N x K, bf16)
// Makes WMMA B-operand loads fully contiguous b128s.
// ---------------------------------------------------------------------------
__global__ void gin_wtrans_kernel(const float* __restrict__ W,
                                  bf16_t* __restrict__ Wt, int K, int N) {
    int i = blockIdx.x * blockDim.x + threadIdx.x;
    if (i >= K * N) return;
    int k = i / N;
    int n = i - k * N;
    Wt[(size_t)n * K + k] = (bf16_t)W[i];
}

// f32 -> bf16 elementwise
__global__ void gin_f2b_kernel(const float* __restrict__ src,
                               bf16_t* __restrict__ dst, int n) {
    int i = blockIdx.x * blockDim.x + threadIdx.x;
    if (i < n) dst[i] = (bf16_t)src[i];
}

// h = (float)x  (init for GIN sum aggregation: h = x + agg)
__global__ void gin_agg_init_kernel(const bf16_t* __restrict__ x,
                                    float* __restrict__ h, int n) {
    int i = blockIdx.x * blockDim.x + threadIdx.x;
    if (i < n) h[i] = (float)x[i];
}

// scatter: h[dst] += x[src], 4 features per thread, f32 global atomics
__global__ void gin_scatter_kernel(const bf16_t* __restrict__ x,
                                   const int* __restrict__ eidx,
                                   float* __restrict__ h,
                                   int D, int nEdges) {
    int t = blockIdx.x * blockDim.x + threadIdx.x;
    int per = D >> 2;
    if (t >= nEdges * per) return;
    int e = t / per;
    int f = (t - e * per) << 2;
    int s = eidx[e];
    int d = eidx[nEdges + e];
    const bf16_t* xs = x + (size_t)s * D + f;
    float* hd = h + (size_t)d * D + f;
    atomicAdd(hd + 0, (float)xs[0]);
    atomicAdd(hd + 1, (float)xs[1]);
    atomicAdd(hd + 2, (float)xs[2]);
    atomicAdd(hd + 3, (float)xs[3]);
}

// ---------------------------------------------------------------------------
// Fused GEMM + bias + ELU (optionally ELU twice):  out = elu^e(A @ W + b)
//   A  : M x K bf16 (row-major)
//   Wt : N x K bf16 (transposed weights, row-major in N)
//   One wave computes a 32x64 strip (2 M-tiles x 4 N-tiles, 8 accumulators).
//   All 12 b128 fragment loads of a K-step are issued before the 8-WMMA
//   burst so the scheduler can use partial s_wait_loadcnt and overlap the
//   tail loads with the first WMMAs. M%32==0, K%32==0, N%64==0 here.
// ---------------------------------------------------------------------------
__global__ void gin_gemm_bf16_kernel(const bf16_t* __restrict__ A,
                                     const bf16_t* __restrict__ Wt,
                                     const float* __restrict__ bias,
                                     float* __restrict__ outF,
                                     bf16_t* __restrict__ outB,
                                     int M, int K, int N, int elu2) {
    const int lane = threadIdx.x & 31;
    const int wave = threadIdx.x >> 5;
    const int gwave = blockIdx.x * (blockDim.x >> 5) + wave;
    const int nGroups = N >> 6;              // 64-wide output groups
    const int mStrips = M >> 5;              // 32-row strips
    if (gwave >= mStrips * nGroups) return;  // wave-uniform: EXEC stays all-1s
    const int mT = gwave / nGroups;
    const int nG = gwave - mT * nGroups;

    const int half = lane >> 4;              // 0: lanes 0-15, 1: lanes 16-31
    const int l16  = lane & 15;

    // Two A-fragment rows (M and M+16); half selects K 0-15 vs 16-31
    const bf16_t* Arow0 = A + (size_t)(mT * 32 + l16)      * K + half * 16;
    const bf16_t* Arow1 = A + (size_t)(mT * 32 + 16 + l16) * K + half * 16;
    // B-fragment rows: column N; half selects K 0-15 vs 16-31
    const bf16_t* Brow0 = Wt + (size_t)(nG * 64 +  0 + l16) * K + half * 16;
    const bf16_t* Brow1 = Wt + (size_t)(nG * 64 + 16 + l16) * K + half * 16;
    const bf16_t* Brow2 = Wt + (size_t)(nG * 64 + 32 + l16) * K + half * 16;
    const bf16_t* Brow3 = Wt + (size_t)(nG * 64 + 48 + l16) * K + half * 16;

    v8f acc[2][4];
    #pragma unroll
    for (int i = 0; i < 2; ++i)
        #pragma unroll
        for (int j = 0; j < 4; ++j) acc[i][j] = (v8f){};

    for (int k = 0; k < K; k += 32) {
        // ---- issue ALL fragment loads for this K-step up front ----
        v8bf a0lo = *(const v8bf*)(Arow0 + k);
        v8bf a0hi = *(const v8bf*)(Arow0 + k + 8);
        v8bf a1lo = *(const v8bf*)(Arow1 + k);
        v8bf a1hi = *(const v8bf*)(Arow1 + k + 8);
        v8bf b0lo = *(const v8bf*)(Brow0 + k);
        v8bf b0hi = *(const v8bf*)(Brow0 + k + 8);
        v8bf b1lo = *(const v8bf*)(Brow1 + k);
        v8bf b1hi = *(const v8bf*)(Brow1 + k + 8);
        v8bf b2lo = *(const v8bf*)(Brow2 + k);
        v8bf b2hi = *(const v8bf*)(Brow2 + k + 8);
        v8bf b3lo = *(const v8bf*)(Brow3 + k);
        v8bf b3hi = *(const v8bf*)(Brow3 + k + 8);
        __builtin_prefetch(Arow0 + k + 32, 0, 1);  // global_prefetch_b8
        __builtin_prefetch(Arow1 + k + 32, 0, 1);

        v16bf a0 = __builtin_shufflevector(a0lo, a0hi,
                     0,1,2,3,4,5,6,7,8,9,10,11,12,13,14,15);
        v16bf a1 = __builtin_shufflevector(a1lo, a1hi,
                     0,1,2,3,4,5,6,7,8,9,10,11,12,13,14,15);
        v16bf b0 = __builtin_shufflevector(b0lo, b0hi,
                     0,1,2,3,4,5,6,7,8,9,10,11,12,13,14,15);
        v16bf b1 = __builtin_shufflevector(b1lo, b1hi,
                     0,1,2,3,4,5,6,7,8,9,10,11,12,13,14,15);
        v16bf b2 = __builtin_shufflevector(b2lo, b2hi,
                     0,1,2,3,4,5,6,7,8,9,10,11,12,13,14,15);
        v16bf b3 = __builtin_shufflevector(b3lo, b3hi,
                     0,1,2,3,4,5,6,7,8,9,10,11,12,13,14,15);

        // ---- 8-WMMA burst ----
        acc[0][0] = __builtin_amdgcn_wmma_f32_16x16x32_bf16(
                        false, a0, false, b0, (short)0, acc[0][0], false, false);
        acc[1][0] = __builtin_amdgcn_wmma_f32_16x16x32_bf16(
                        false, a1, false, b0, (short)0, acc[1][0], false, false);
        acc[0][1] = __builtin_amdgcn_wmma_f32_16x16x32_bf16(
                        false, a0, false, b1, (short)0, acc[0][1], false, false);
        acc[1][1] = __builtin_amdgcn_wmma_f32_16x16x32_bf16(
                        false, a1, false, b1, (short)0, acc[1][1], false, false);
        acc[0][2] = __builtin_amdgcn_wmma_f32_16x16x32_bf16(
                        false, a0, false, b2, (short)0, acc[0][2], false, false);
        acc[1][2] = __builtin_amdgcn_wmma_f32_16x16x32_bf16(
                        false, a1, false, b2, (short)0, acc[1][2], false, false);
        acc[0][3] = __builtin_amdgcn_wmma_f32_16x16x32_bf16(
                        false, a0, false, b3, (short)0, acc[0][3], false, false);
        acc[1][3] = __builtin_amdgcn_wmma_f32_16x16x32_bf16(
                        false, a1, false, b3, (short)0, acc[1][3], false, false);
    }

    // Epilogue: C/D layout §7.12.2 — VGPR r: lanes0-15 M=r, lanes16-31 M=r+8
    #pragma unroll
    for (int j = 0; j < 4; ++j) {
        const int col = nG * 64 + j * 16 + l16;
        const float bj = bias[col];
        #pragma unroll
        for (int i = 0; i < 2; ++i) {
            #pragma unroll
            for (int r = 0; r < 8; ++r) {
                float v = acc[i][j][r] + bj;
                v = elu1(v);
                if (elu2) v = elu1(v);
                const size_t row = (size_t)(mT * 32 + i * 16 + half * 8 + r);
                if (outF) outF[row * N + col] = v;
                if (outB) outB[row * N + col] = (bf16_t)v;
            }
        }
    }
}

// ---------------------------------------------------------------------------
// Host orchestration
// ---------------------------------------------------------------------------
static inline void launch_gemm(const bf16_t* A, const bf16_t* Wt, const float* b,
                               float* outF, bf16_t* outB,
                               int M, int K, int N, int elu2, hipStream_t s) {
    int waves = (M >> 5) * (N >> 6);
    int blocks = (waves + 7) / 8;            // 8 waves (256 threads) per block
    gin_gemm_bf16_kernel<<<blocks, 256, 0, s>>>(A, Wt, b, outF, outB,
                                                M, K, N, elu2);
}

extern "C" void kernel_launch(void* const* d_in, const int* in_sizes, int n_in,
                              void* d_out, int out_size, void* d_ws, size_t ws_size,
                              hipStream_t stream) {
    const int NN = 20000, NE = 320000, IN = 256, H1 = 640, H2 = 320, OUT = 256;

    const float* features = (const float*)d_in[0];
    const int*   edges    = (const int*)d_in[1];
    const float* w1a = (const float*)d_in[2];  const float* b1a = (const float*)d_in[3];
    const float* w1b = (const float*)d_in[4];  const float* b1b = (const float*)d_in[5];
    const float* w2a = (const float*)d_in[6];  const float* b2a = (const float*)d_in[7];
    const float* w2b = (const float*)d_in[8];  const float* b2b = (const float*)d_in[9];
    const float* w3a = (const float*)d_in[10]; const float* b3a = (const float*)d_in[11];
    const float* w3b = (const float*)d_in[12]; const float* b3b = (const float*)d_in[13];
    const float* wr  = (const float*)d_in[14]; const float* br  = (const float*)d_in[15];

    float* outX   = (float*)d_out;                       // x  : 20000 x 256
    float* outRes = (float*)d_out + (size_t)NN * OUT;    // res: 20000 x 256

    // ---- carve workspace (256B aligned chunks) ----
    char* wp = (char*)d_ws;
    auto carve = [&](size_t bytes) -> void* {
        void* p = (void*)wp;
        wp += (bytes + 255) & ~(size_t)255;
        return p;
    };
    float*  hf = (float*)carve((size_t)NN * H1 * sizeof(float));   // agg scratch
    bf16_t* B0 = (bf16_t*)carve((size_t)NN * H1 * sizeof(bf16_t)); // rotating bf16
    bf16_t* B1 = (bf16_t*)carve((size_t)NN * H1 * sizeof(bf16_t));
    bf16_t* B2 = (bf16_t*)carve((size_t)NN * H1 * sizeof(bf16_t));
    bf16_t* wrT  = (bf16_t*)carve((size_t)IN * OUT * sizeof(bf16_t));
    bf16_t* w1aT = (bf16_t*)carve((size_t)IN * H1  * sizeof(bf16_t));
    bf16_t* w1bT = (bf16_t*)carve((size_t)H1 * H1  * sizeof(bf16_t));
    bf16_t* w2aT = (bf16_t*)carve((size_t)H1 * H2  * sizeof(bf16_t));
    bf16_t* w2bT = (bf16_t*)carve((size_t)H2 * H2  * sizeof(bf16_t));
    bf16_t* w3aT = (bf16_t*)carve((size_t)H2 * OUT * sizeof(bf16_t));
    bf16_t* w3bT = (bf16_t*)carve((size_t)OUT * OUT * sizeof(bf16_t));

    auto gridFor = [](int n) { return (n + 255) / 256; };

    // ---- weights: transpose + bf16 ----
    gin_wtrans_kernel<<<gridFor(IN*OUT), 256, 0, stream>>>(wr,  wrT,  IN,  OUT);
    gin_wtrans_kernel<<<gridFor(IN*H1),  256, 0, stream>>>(w1a, w1aT, IN,  H1);
    gin_wtrans_kernel<<<gridFor(H1*H1),  256, 0, stream>>>(w1b, w1bT, H1,  H1);
    gin_wtrans_kernel<<<gridFor(H1*H2),  256, 0, stream>>>(w2a, w2aT, H1,  H2);
    gin_wtrans_kernel<<<gridFor(H2*H2),  256, 0, stream>>>(w2b, w2bT, H2,  H2);
    gin_wtrans_kernel<<<gridFor(H2*OUT), 256, 0, stream>>>(w3a, w3aT, H2,  OUT);
    gin_wtrans_kernel<<<gridFor(OUT*OUT),256, 0, stream>>>(w3b, w3bT, OUT, OUT);

    // ---- features -> bf16 (B0) ----
    gin_f2b_kernel<<<gridFor(NN*IN), 256, 0, stream>>>(features, B0, NN*IN);

    // ---- residual: res = elu(features @ wr + br) ----
    launch_gemm(B0, wrT, br, outRes, nullptr, NN, IN, OUT, 0, stream);

    // ---- Layer 1 (x = B0, D=256) ----
    gin_agg_init_kernel<<<gridFor(NN*IN), 256, 0, stream>>>(B0, hf, NN*IN);
    gin_scatter_kernel<<<gridFor(NE*(IN/4)), 256, 0, stream>>>(B0, edges, hf, IN, NE);
    gin_f2b_kernel<<<gridFor(NN*IN), 256, 0, stream>>>(hf, B1, NN*IN);
    launch_gemm(B1, w1aT, b1a, nullptr, B2, NN, IN, H1, 0, stream);   // elu
    launch_gemm(B2, w1bT, b1b, nullptr, B0, NN, H1, H1, 1, stream);   // elu(elu)

    // ---- Layer 2 (x = B0, D=640) ----
    gin_agg_init_kernel<<<gridFor(NN*H1), 256, 0, stream>>>(B0, hf, NN*H1);
    gin_scatter_kernel<<<gridFor(NE*(H1/4)), 256, 0, stream>>>(B0, edges, hf, H1, NE);
    gin_f2b_kernel<<<gridFor(NN*H1), 256, 0, stream>>>(hf, B1, NN*H1);
    launch_gemm(B1, w2aT, b2a, nullptr, B2, NN, H1, H2, 0, stream);   // elu
    launch_gemm(B2, w2bT, b2b, nullptr, B0, NN, H2, H2, 1, stream);   // elu(elu)

    // ---- Layer 3 (x = B0, D=320) ----
    gin_agg_init_kernel<<<gridFor(NN*H2), 256, 0, stream>>>(B0, hf, NN*H2);
    gin_scatter_kernel<<<gridFor(NE*(H2/4)), 256, 0, stream>>>(B0, edges, hf, H2, NE);
    gin_f2b_kernel<<<gridFor(NN*H2), 256, 0, stream>>>(hf, B1, NN*H2);
    launch_gemm(B1, w3aT, b3a, nullptr, B2, NN, H2, OUT, 0, stream);  // elu
    launch_gemm(B2, w3bT, b3b, outX, nullptr, NN, OUT, OUT, 0, stream); // elu, f32 out
}
